// stCAMBL_Lib_9139690405917
// MI455X (gfx1250) — matrix operand order repaired and checked by reading, not compiled, not run on hardware
//
#include <hip/hip_runtime.h>
#include <hip/hip_bf16.h>
#include <math.h>

// ---------------------------------------------------------------------------
// stCAMBL forward on MI455X (gfx1250).
//  * All GEMMs -> v_wmma_f32_16x16x32_bf16 (f32 accumulate).
//  * adj converted once to bf16 (134MB -> fits 192MB L2) so the dominant
//    de_feat GEMM (adj @ [8192,3008], ~402 GFLOP) streams A from L2.
//  * adj-GEMMs stage their A tile with the Tensor Data Mover
//    (tensor_load_to_lds + s_wait_tensorcnt); everything else uses vector
//    b128 staging. All operands padded in ws so staging is branchless.
// ---------------------------------------------------------------------------

#define NN   8192
#define GG   3000
#define GGP  3008          // GG padded to K-multiple of 32 / N-multiple of 64
#define FH1  64
#define FH2  32
#define GH1  64
#define GH2  32
#define LATENT 64          // FH2 + GH2
#define NCLUST 10
#define INV_BN 0.9995003746877732f   // 1/sqrt(1+1e-3)

typedef __attribute__((ext_vector_type(16))) __bf16 v16bf;
typedef __attribute__((ext_vector_type(8)))  float  v8f;
typedef __attribute__((ext_vector_type(4)))  unsigned int u32x4;
typedef __attribute__((ext_vector_type(8)))  int    i32x8;
typedef __attribute__((ext_vector_type(4)))  int    i32x4;

#define BM 128
#define BN 64
#define BK 32

// ------------------------------ fragments ----------------------------------

__device__ inline v16bf load_frag_A(const __bf16* As, int mbase, int lane) {
    // 16-bit A 16x32: lane<16 -> K {0..7,16..23}; lane>=16 -> K {8..15,24..31}
    int m = mbase + (lane & 15);
    int khalf = lane >> 4;
    const __bf16* p = As + m * BK + khalf * 8;
    union { v16bf v; uint4 q[2]; } u;
    u.q[0] = *(const uint4*)(p);
    u.q[1] = *(const uint4*)(p + 16);
    return u.v;
}

__device__ inline v16bf load_frag_B(const __bf16* Bt, int nbase, int lane) {
    // 16-bit B 32x16: lane<16 -> K {0..7,16..23}; lane>=16 -> K {8..15,24..31} (col=lane%16)
    int n = nbase + (lane & 15);
    int khalf = lane >> 4;
    const __bf16* p = Bt + n * BK + khalf * 8;
    union { v16bf v; uint4 q[2]; } u;
    u.q[0] = *(const uint4*)(p);
    u.q[1] = *(const uint4*)(p + 16);
    return u.v;
}

// ----------------------------- staging helpers -----------------------------
// Branchless: callers guarantee padded allocations
//   A: M % 128 == 0, K % 32 == 0, row stride lda >= K
//   B: allocated columns >= gridDim.x*64, K rows

__device__ inline void stage_A(const __bf16* __restrict__ A, int lda,
                               int m0, int k0, int tid, __bf16* As)
{
    int idx = tid * 16;            // 128x32 tile, 16 bf16 per thread
    int r = idx >> 5;              // row 0..127
    int c = idx & 31;              // 0 or 16
    const uint4* src = (const uint4*)(A + (long)(m0 + r) * lda + (k0 + c));
    *(uint4*)(As + idx)     = src[0];
    *(uint4*)(As + idx + 8) = src[1];
}

__device__ inline void stage_B(const __bf16* __restrict__ B, int ldb,
                               int k0, int n0, int tid, __bf16* Bt)
{
    // 32x64 tile; each thread: 2 k-rows x 4 n-cols, packed b32 stores into
    // transposed LDS Bt[n][k] (k-pairs contiguous).
    int c = (tid & 15) * 4;        // n offset 0..60
    int r = (tid >> 4) * 2;        // k offset 0,2,..,30
    const __bf16* src = B + (long)(k0 + r) * ldb + (n0 + c);
    union { uint2 u; unsigned short s[4]; } k0row, k1row;
    k0row.u = *(const uint2*)(src);
    k1row.u = *(const uint2*)(src + ldb);
    for (int i = 0; i < 4; ++i) {
        unsigned v = (unsigned)k0row.s[i] | ((unsigned)k1row.s[i] << 16);
        *(unsigned*)(Bt + (c + i) * BK + r) = v;
    }
}

// ------------------------------ epilogue -----------------------------------

__device__ inline void epilogue_store(v8f accs[2][2], int m0, int n0,
                                      int wm, int wn, int lane,
                                      float* __restrict__ Cf, int ldcf,
                                      __bf16* __restrict__ Cb, int ldcb,
                                      int N, int Nb, int epi,
                                      const float* __restrict__ bias,
                                      const float* __restrict__ gamma,
                                      const float* __restrict__ beta)
{
    const int rowBase = m0 + 32 * wm;
    const int colBase = n0 + 32 * wn;
    for (int i = 0; i < 2; ++i) {
        for (int j = 0; j < 2; ++j) {
            int col = colBase + 16 * j + (lane & 15);
            int rb  = rowBase + 16 * i + ((lane >> 4) ? 8 : 0);
            float bv = 0.f, gv = 1.f, bev = 0.f;
            if (epi == 2 && col < N) { bv = bias[col]; gv = gamma[col]; bev = beta[col]; }
            v8f a = accs[i][j];
            for (int r = 0; r < 8; ++r) {
                int row = rb + r;
                float v = a[r];
                if (epi == 1) {                       // ReLU
                    v = v > 0.f ? v : 0.f;
                } else if (epi == 2) {                // bias + BN(eval) + ELU
                    v = (v + bv) * INV_BN * gv + bev;
                    v = v > 0.f ? v : (__expf(v) - 1.f);
                }
                if (Cf && col < N)  Cf[(long)row * ldcf + col] = v;
                if (Cb && col < Nb) Cb[(long)row * ldcb + col] =
                        (col < N) ? (__bf16)v : (__bf16)0.0f;   // maintain zero pad
            }
        }
    }
}

// --------------------------- WMMA GEMM (vector) ----------------------------

__global__ __launch_bounds__(256)
void gemm_bf16_kernel(const __bf16* __restrict__ A, int lda,
                      const __bf16* __restrict__ B, int ldb,
                      float* __restrict__ Cf, int ldcf,
                      __bf16* __restrict__ Cb, int ldcb,
                      int M, int N, int Nb, int K, int epi,
                      const float* __restrict__ bias,
                      const float* __restrict__ gamma,
                      const float* __restrict__ beta)
{
    __shared__ __align__(16) __bf16 As[BM * BK];
    __shared__ __align__(16) __bf16 Bt[BN * BK];

    const int tid  = threadIdx.x;
    const int lane = tid & 31;
    const int wave = tid >> 5;
    const int wm   = wave >> 1;          // 0..3
    const int wn   = wave & 1;           // 0..1
    const int m0 = blockIdx.y * BM;
    const int n0 = blockIdx.x * BN;

    v8f c00 = {}, c01 = {}, c10 = {}, c11 = {};

    for (int k0 = 0; k0 < K; k0 += BK) {
        stage_A(A, lda, m0, k0, tid, As);
        stage_B(B, ldb, k0, n0, tid, Bt);
        if (k0 + BK < K)                  // global_prefetch_b8 of next A tile
            __builtin_prefetch(A + (long)(m0 + (tid >> 1)) * lda + k0 + BK, 0, 1);
        __syncthreads();

        v16bf a0 = load_frag_A(As, 32 * wm,      lane);
        v16bf a1 = load_frag_A(As, 32 * wm + 16, lane);
        v16bf b0 = load_frag_B(Bt, 32 * wn,      lane);
        v16bf b1 = load_frag_B(Bt, 32 * wn + 16, lane);
        c00 = __builtin_amdgcn_wmma_f32_16x16x32_bf16(false, a0, false, b0, (short)0, c00, false, false);
        c01 = __builtin_amdgcn_wmma_f32_16x16x32_bf16(false, a0, false, b1, (short)0, c01, false, false);
        c10 = __builtin_amdgcn_wmma_f32_16x16x32_bf16(false, a1, false, b0, (short)0, c10, false, false);
        c11 = __builtin_amdgcn_wmma_f32_16x16x32_bf16(false, a1, false, b1, (short)0, c11, false, false);
        __syncthreads();
    }

    v8f accs[2][2] = { { c00, c01 }, { c10, c11 } };
    epilogue_store(accs, m0, n0, wm, wn, lane, Cf, ldcf, Cb, ldcb,
                   N, Nb, epi, bias, gamma, beta);
}

// ----------------------- WMMA GEMM (TDM A staging) -------------------------
// A tile (128x32 bf16) fetched by the Tensor Data Mover. D# built per
// ISA ch.8: group0 {count=1, lds_addr, global_addr, type=2},
// group1 {data_size=2B, tensor_dim0=K, tensor_dim1=M, tile=32x128,
//          tensor_dim0_stride=lda}.

__global__ __launch_bounds__(256)
void gemm_bf16_tdm_kernel(const __bf16* __restrict__ A, int lda,
                          const __bf16* __restrict__ B, int ldb,
                          float* __restrict__ Cf, int ldcf,
                          __bf16* __restrict__ Cb, int ldcb,
                          int M, int N, int Nb, int K, int epi,
                          const float* __restrict__ bias,
                          const float* __restrict__ gamma,
                          const float* __restrict__ beta)
{
    __shared__ __align__(16) __bf16 As[BM * BK];
    __shared__ __align__(16) __bf16 Bt[BN * BK];

    const int tid  = threadIdx.x;
    const int lane = tid & 31;
    const int wave = tid >> 5;
    const int wm   = wave >> 1;
    const int wn   = wave & 1;
    const int m0 = blockIdx.y * BM;
    const int n0 = blockIdx.x * BN;

    const unsigned long long abase =
        (unsigned long long)(size_t)(A + (long)m0 * lda);     // byte address
    const unsigned ldsA = (unsigned)(size_t)&As[0];           // LDS byte offset

    // constant D# words
    i32x8 g1c;
    g1c[0] = (1 << 16);                                       // data_size = 2B
    g1c[1] = (int)(((unsigned)K & 0xFFFFu) << 16);            // tensor_dim0 lo
    g1c[2] = (int)((((unsigned)K >> 16) & 0xFFFFu)
                 | (((unsigned)M & 0xFFFFu) << 16));          // dim0 hi | dim1 lo
    g1c[3] = (int)((((unsigned)M >> 16) & 0xFFFFu)
                 | (32u << 16));                              // dim1 hi | tile_dim0=32
    g1c[4] = 128;                                             // tile_dim1=128
    g1c[5] = lda;                                             // dim0_stride lo32
    g1c[6] = 0;                                               // dim0_stride[47:32]=0
    g1c[7] = 0;
    const i32x4 gz4 = { 0, 0, 0, 0 };
    const i32x8 gz8 = { 0, 0, 0, 0, 0, 0, 0, 0 };

    v8f c00 = {}, c01 = {}, c10 = {}, c11 = {};

    for (int k0 = 0; k0 < K; k0 += BK) {
        stage_B(B, ldb, k0, n0, tid, Bt);
        if (wave == 0) {
            unsigned long long ga = abase + (unsigned long long)k0 * 2ull;
            u32x4 g0;
            g0[0] = 1u;                                       // count=1 (valid D#)
            g0[1] = ldsA;                                     // lds_addr
            g0[2] = (unsigned)(ga & 0xFFFFFFFFull);           // global_addr lo
            g0[3] = (unsigned)((ga >> 32) & 0x01FFFFFFull)
                  | (2u << 30);                               // addr hi | type=2
            __builtin_amdgcn_tensor_load_to_lds(g0, g1c, gz4, gz4, gz8, 0);
            __builtin_amdgcn_s_wait_tensorcnt(0);
        }
        __syncthreads();

        v16bf a0 = load_frag_A(As, 32 * wm,      lane);
        v16bf a1 = load_frag_A(As, 32 * wm + 16, lane);
        v16bf b0 = load_frag_B(Bt, 32 * wn,      lane);
        v16bf b1 = load_frag_B(Bt, 32 * wn + 16, lane);
        c00 = __builtin_amdgcn_wmma_f32_16x16x32_bf16(false, a0, false, b0, (short)0, c00, false, false);
        c01 = __builtin_amdgcn_wmma_f32_16x16x32_bf16(false, a0, false, b1, (short)0, c01, false, false);
        c10 = __builtin_amdgcn_wmma_f32_16x16x32_bf16(false, a1, false, b0, (short)0, c10, false, false);
        c11 = __builtin_amdgcn_wmma_f32_16x16x32_bf16(false, a1, false, b1, (short)0, c11, false, false);
        __syncthreads();
    }

    v8f accs[2][2] = { { c00, c01 }, { c10, c11 } };
    epilogue_store(accs, m0, n0, wm, wn, lane, Cf, ldcf, Cb, ldcb,
                   N, Nb, epi, bias, gamma, beta);
}

// --------------------------- elementwise kernels ---------------------------

__device__ inline unsigned hash_u32(unsigned x) {
    x ^= x >> 16; x *= 0x7feb352dU; x ^= x >> 15; x *= 0x846ca68bU; x ^= x >> 16;
    return x;
}

// Deterministic stand-in for the threefry mask (same statistics: 70% nodes
// masked; bands [0,49)/[49,99)/[99,199) un-mask at 10%/5%/1%).
__device__ inline float mask_val(int i, int j) {
    unsigned hn = hash_u32((unsigned)i * 2654435761u + 12345u);
    if ((hn % 10u) >= 7u) return 0.0f;
    float p = (j < 49) ? 0.1f : (j < 99) ? 0.05f : (j < 199) ? 0.01f : 0.0f;
    if (p > 0.0f) {
        unsigned hc = hash_u32((unsigned)(i * 3001 + j) ^ 0x9e3779b9u);
        if ((float)(hc & 0xFFFFu) * (1.0f / 65536.0f) < p) return 0.0f;
    }
    return 1.0f;
}

__global__ void mask_xm_kernel(const float* __restrict__ x,
                               const float* __restrict__ mt,
                               __bf16* __restrict__ xm_bf)   // [NN][GGP]
{
    long total = (long)NN * GGP;
    for (long idx = (long)blockIdx.x * blockDim.x + threadIdx.x; idx < total;
         idx += (long)gridDim.x * blockDim.x) {
        int i = (int)(idx / GGP), j = (int)(idx % GGP);
        float v = 0.0f;
        if (j < GG) {
            long s = (long)i * GG + j;
            v = x[s] + mask_val(i, j) * mt[s];
        }
        xm_bf[idx] = (__bf16)v;
    }
}

// dst [Rp][ldd] = bf16(src[R][C]) zero-padded
__global__ void convert_pad_bf16_kernel(const float* __restrict__ src, int R, int C,
                                        __bf16* __restrict__ dst, int Rp, int ldd)
{
    long total = (long)Rp * ldd;
    for (long idx = (long)blockIdx.x * blockDim.x + threadIdx.x; idx < total;
         idx += (long)gridDim.x * blockDim.x) {
        int r = (int)(idx / ldd), c = (int)(idx % ldd);
        float v = (r < R && c < C) ? src[(long)r * C + c] : 0.0f;
        dst[idx] = (__bf16)v;
    }
}

__global__ void concat_z_kernel(const float* __restrict__ feat,
                                const float* __restrict__ mu,
                                float* __restrict__ z,
                                __bf16* __restrict__ zb)
{
    long total = (long)NN * LATENT;
    for (long idx = (long)blockIdx.x * blockDim.x + threadIdx.x; idx < total;
         idx += (long)gridDim.x * blockDim.x) {
        int row = (int)(idx / LATENT), c = (int)(idx % LATENT);
        float v = (c < FH2) ? feat[row * FH2 + c] : mu[row * GH2 + (c - FH2)];
        z[idx] = v;
        zb[idx] = (__bf16)v;
    }
}

__global__ void q_kernel(const float* __restrict__ z,
                         const float* __restrict__ cluster,
                         float* __restrict__ q)
{
    int row = blockIdx.x * blockDim.x + threadIdx.x;
    if (row >= NN) return;
    float qs[NCLUST]; float s = 0.f;
    for (int c = 0; c < NCLUST; ++c) {
        float d2 = 0.f;
        for (int k = 0; k < LATENT; ++k) {
            float d = z[row * LATENT + k] - cluster[c * LATENT + k];
            d2 += d * d;
        }
        float v = 1.0f / (1.0f + d2);       // alpha=1 -> exponent 1
        qs[c] = v; s += v;
    }
    float inv = 1.0f / s;
    for (int c = 0; c < NCLUST; ++c) q[row * NCLUST + c] = qs[c] * inv;
}

__global__ void copy_f32_kernel(const float* __restrict__ s, float* __restrict__ d, long n)
{
    for (long i = (long)blockIdx.x * blockDim.x + threadIdx.x; i < n;
         i += (long)gridDim.x * blockDim.x)
        d[i] = s[i];
}

__global__ void zero_f32_kernel(float* p, long n)
{
    for (long i = (long)blockIdx.x * blockDim.x + threadIdx.x; i < n;
         i += (long)gridDim.x * blockDim.x)
        p[i] = 0.0f;
}

__global__ void loss_kernel(const float* __restrict__ x,
                            const float* __restrict__ mt,
                            const float* __restrict__ de,
                            float* __restrict__ out)
{
    __shared__ float sm[256];
    long total = (long)NN * GG;
    float acc = 0.f;
    for (long idx = (long)blockIdx.x * blockDim.x + threadIdx.x; idx < total;
         idx += (long)gridDim.x * blockDim.x) {
        int i = (int)(idx / GG), j = (int)(idx % GG);
        float m  = mask_val(i, j);
        float xm = x[idx] + m * mt[idx];
        float d  = (de[idx] - xm) * m;
        acc += d * d;
    }
    sm[threadIdx.x] = acc;
    __syncthreads();
    for (int s = 128; s > 0; s >>= 1) {
        if (threadIdx.x < (unsigned)s) sm[threadIdx.x] += sm[threadIdx.x + s];
        __syncthreads();
    }
    if (threadIdx.x == 0) atomicAdd(out, sm[0] / (float)total);
}

// ------------------------------- launcher ----------------------------------

static void launch_gemm(bool tdm,
                        const __bf16* A, int lda, const __bf16* B, int ldb,
                        float* Cf, int ldcf, __bf16* Cb, int ldcb,
                        int M, int N, int Nb, int K, int epi,
                        const float* b, const float* g, const float* be,
                        hipStream_t stream)
{
    int ncols = (N > Nb) ? N : Nb;
    dim3 grid((ncols + BN - 1) / BN, M / BM);
    if (tdm)
        gemm_bf16_tdm_kernel<<<grid, dim3(256), 0, stream>>>(A, lda, B, ldb,
            Cf, ldcf, Cb, ldcb, M, N, Nb, K, epi, b, g, be);
    else
        gemm_bf16_kernel<<<grid, dim3(256), 0, stream>>>(A, lda, B, ldb,
            Cf, ldcf, Cb, ldcb, M, N, Nb, K, epi, b, g, be);
}

static inline size_t align256(size_t x) { return (x + 255) & ~(size_t)255; }

extern "C" void kernel_launch(void* const* d_in, const int* in_sizes, int n_in,
                              void* d_out, int out_size, void* d_ws, size_t ws_size,
                              hipStream_t stream)
{
    (void)in_sizes; (void)n_in; (void)out_size; (void)ws_size;

    const float* x    = (const float*)d_in[0];
    const float* adj  = (const float*)d_in[1];
    const float* W1   = (const float*)d_in[2];
    const float* b1   = (const float*)d_in[3];
    const float* g1   = (const float*)d_in[4];
    const float* be1  = (const float*)d_in[5];
    const float* W2   = (const float*)d_in[6];
    const float* b2   = (const float*)d_in[7];
    const float* g2   = (const float*)d_in[8];
    const float* be2  = (const float*)d_in[9];
    const float* gc1w = (const float*)d_in[10];
    const float* gc2w = (const float*)d_in[11];
    const float* gc3w = (const float*)d_in[12];
    const float* decw = (const float*)d_in[13];
    const float* clus = (const float*)d_in[14];
    const float* mtok = (const float*)d_in[15];

    float* out = (float*)d_out;
    const long z_off      = 0;
    const long mu_off     = z_off      + (long)NN * LATENT;
    const long logvar_off = mu_off     + (long)NN * GH2;
    const long defeat_off = logvar_off + (long)NN * GH2;
    const long q_off      = defeat_off + (long)NN * GG;
    const long feat_off   = q_off      + (long)NN * NCLUST;
    const long gnnz_off   = feat_off   + (long)NN * FH2;
    const long loss_off   = gnnz_off   + (long)NN * GH2;

    char* ws = (char*)d_ws; size_t off = 0;
    auto alloc = [&](size_t bytes) { char* p = ws + off; off += align256(bytes); return p; };
    __bf16* adj_bf  = (__bf16*)alloc((size_t)NN * NN * 2);     // [8192][8192]
    __bf16* xm_bf   = (__bf16*)alloc((size_t)NN * GGP * 2);    // [8192][3008]
    __bf16* Y_bf    = (__bf16*)alloc((size_t)NN * GGP * 2);    // [8192][3008]
    __bf16* w1p     = (__bf16*)alloc((size_t)GGP * FH1 * 2);   // [3008][64]
    __bf16* w2p     = (__bf16*)alloc((size_t)FH1 * 64 * 2);    // [64][64]
    __bf16* gc1wp   = (__bf16*)alloc((size_t)FH2 * GH1 * 2);   // [32][64]
    __bf16* gc2wp   = (__bf16*)alloc((size_t)GH1 * 64 * 2);    // [64][64]
    __bf16* gc3wp   = (__bf16*)alloc((size_t)GH1 * 64 * 2);    // [64][64]
    __bf16* decwp   = (__bf16*)alloc((size_t)LATENT * GGP * 2);// [64][3008]
    __bf16* h1e_bf  = (__bf16*)alloc((size_t)NN * FH1 * 2);    // [8192][64]
    __bf16* feat_bf = (__bf16*)alloc((size_t)NN * 64 * 2);     // [8192][64]
    __bf16* t1_bf   = (__bf16*)alloc((size_t)NN * GH1 * 2);    // [8192][64]
    __bf16* h1_bf   = (__bf16*)alloc((size_t)NN * GH1 * 2);    // [8192][64]
    __bf16* t2_bf   = (__bf16*)alloc((size_t)NN * 64 * 2);     // [8192][64]
    __bf16* t3_bf   = (__bf16*)alloc((size_t)NN * 64 * 2);     // [8192][64]
    __bf16* z_bf    = (__bf16*)alloc((size_t)NN * LATENT * 2); // [8192][64]

    const int T = 256;
    auto gs = [&](long n) { long b = (n + T - 1) / T; return (int)(b > 4096 ? 4096 : b); };

    // ---- conversions (adj once -> bf16: L2-resident afterwards) ----
    convert_pad_bf16_kernel<<<gs((long)NN * NN), T, 0, stream>>>(adj, NN, NN, adj_bf, NN, NN);
    convert_pad_bf16_kernel<<<gs((long)GGP * FH1), T, 0, stream>>>(W1, GG, FH1, w1p, GGP, FH1);
    convert_pad_bf16_kernel<<<gs(FH1 * 64), T, 0, stream>>>(W2, FH1, FH2, w2p, FH1, 64);
    convert_pad_bf16_kernel<<<gs(FH2 * GH1), T, 0, stream>>>(gc1w, FH2, GH1, gc1wp, FH2, GH1);
    convert_pad_bf16_kernel<<<gs(GH1 * 64), T, 0, stream>>>(gc2w, GH1, GH2, gc2wp, GH1, 64);
    convert_pad_bf16_kernel<<<gs(GH1 * 64), T, 0, stream>>>(gc3w, GH1, GH2, gc3wp, GH1, 64);
    convert_pad_bf16_kernel<<<gs((long)LATENT * GGP), T, 0, stream>>>(decw, LATENT, GG, decwp, LATENT, GGP);

    // ---- mask + xm (padded [8192][3008]) ----
    mask_xm_kernel<<<gs((long)NN * GGP), T, 0, stream>>>(x, mtok, xm_bf);

    // ---- encoder: feat = elu(bn(elu(bn(xm@W1+b1))@W2+b2)) ----
    launch_gemm(false, xm_bf, GGP, w1p, FH1, nullptr, 0, h1e_bf, FH1,
                NN, FH1, FH1, GGP, 2, b1, g1, be1, stream);
    launch_gemm(false, h1e_bf, FH1, w2p, 64, out + feat_off, FH2, feat_bf, 64,
                NN, FH2, 64, FH1, 2, b2, g2, be2, stream);

    // ---- GCN: h1 = relu(adj @ (feat @ gc1_w)) ----
    launch_gemm(false, feat_bf, 64, gc1wp, GH1, nullptr, 0, t1_bf, GH1,
                NN, GH1, GH1, FH2, 0, nullptr, nullptr, nullptr, stream);
    launch_gemm(true,  adj_bf, NN, t1_bf, GH1, nullptr, 0, h1_bf, GH1,
                NN, GH1, GH1, NN, 1, nullptr, nullptr, nullptr, stream);

    // ---- mu = adj @ (h1 @ gc2_w) ; logvar = adj @ (h1 @ gc3_w) ----
    launch_gemm(false, h1_bf, GH1, gc2wp, 64, nullptr, 0, t2_bf, 64,
                NN, GH2, 64, GH1, 0, nullptr, nullptr, nullptr, stream);
    launch_gemm(false, h1_bf, GH1, gc3wp, 64, nullptr, 0, t3_bf, 64,
                NN, GH2, 64, GH1, 0, nullptr, nullptr, nullptr, stream);
    launch_gemm(true,  adj_bf, NN, t2_bf, 64, out + mu_off, GH2, nullptr, 0,
                NN, GH2, 0, NN, 0, nullptr, nullptr, nullptr, stream);
    launch_gemm(true,  adj_bf, NN, t3_bf, 64, out + logvar_off, GH2, nullptr, 0,
                NN, GH2, 0, NN, 0, nullptr, nullptr, nullptr, stream);

    // ---- z = [feat, mu] ; gnn_z = mu ----
    concat_z_kernel<<<gs((long)NN * LATENT), T, 0, stream>>>(out + feat_off, out + mu_off,
                                                             out + z_off, z_bf);
    copy_f32_kernel<<<gs((long)NN * GH2), T, 0, stream>>>(out + mu_off, out + gnnz_off,
                                                          (long)NN * GH2);

    // ---- decoder: de_feat = adj @ (z @ dec_w)  (dominant ~402 GFLOP) ----
    launch_gemm(false, z_bf, LATENT, decwp, GGP, nullptr, 0, Y_bf, GGP,
                NN, GG, GGP, LATENT, 0, nullptr, nullptr, nullptr, stream);
    launch_gemm(true,  adj_bf, NN, Y_bf, GGP, out + defeat_off, GG, nullptr, 0,
                NN, GG, 0, NN, 0, nullptr, nullptr, nullptr, stream);

    // ---- DEC soft assignment q ----
    q_kernel<<<(NN + T - 1) / T, T, 0, stream>>>(out + z_off, clus, out + q_off);

    // ---- masked MSE loss ----
    zero_f32_kernel<<<1, 1, 0, stream>>>(out + loss_off, 1);
    loss_kernel<<<2048, T, 0, stream>>>(x, mtok, out + defeat_off, out + loss_off);
}
